// SingleMambaBlock_2_24223615549513
// MI455X (gfx1250) — matrix operand order repaired and checked by reading, not compiled, not run on hardware
//
#include <hip/hip_runtime.h>
#include <hip/hip_bf16.h>
#include <math.h>

typedef _Float16 v16h __attribute__((ext_vector_type(16)));
typedef _Float16 v8h  __attribute__((ext_vector_type(8)));
typedef float    v8f  __attribute__((ext_vector_type(8)));

#define DMODEL 96
#define DSTATE 8
#define DCONV  4
#define DTRANK 6
#define XDIM   (DTRANK + 2*DSTATE)   // 22
#define NDBL   (2*DMODEL)            // 192

__device__ __forceinline__ float sigmoidf_(float x){ return 1.f/(1.f+__expf(-x)); }
__device__ __forceinline__ float softplusf_(float x){ return x > 20.f ? x : log1pf(__expf(x)); }

// ---------------- LayerNorm + direction-transpose, f16 output ----------------
__global__ void k_ln_transpose(const float* __restrict__ x, const float* __restrict__ gam,
                               const float* __restrict__ bet, _Float16* __restrict__ o16,
                               int mode, int Cc, int Hc, int Wc, int L)
{
    __shared__ float s1[128], s2[128];
    int row = blockIdx.x;             // b*L + l
    int b = row / L, l = row % L;
    size_t base; int stride;
    if (mode == 1)      { int h = l / Wc, w = l % Wc; base = ((size_t)b*Cc*Hc + h)*Wc + w;           stride = Hc*Wc; }
    else if (mode == 2) { int c = l / Wc, w = l % Wc; base = (((size_t)b*Cc + c)*Hc)*Wc + w;         stride = Wc;    }
    else                { int c = l / Hc, h = l % Hc; base = (((size_t)b*Cc + c)*Hc + h)*(size_t)Wc; stride = 1;     }
    int t = threadIdx.x;
    float v = (t < DMODEL) ? x[base + (size_t)t*stride] : 0.f;
    s1[t] = v; s2[t] = v*v;
    __syncthreads();
    for (int off = 64; off > 0; off >>= 1) {
        if (t < off) { s1[t] += s1[t+off]; s2[t] += s2[t+off]; }
        __syncthreads();
    }
    float mu  = s1[0] * (1.f/DMODEL);
    float var = s2[0] * (1.f/DMODEL) - mu*mu;
    float inv = rsqrtf(var + 1e-5f);
    if (t < DMODEL) {
        float o = (v - mu) * inv * gam[t] + bet[t];
        o16[(size_t)row*DMODEL + t] = (_Float16)o;
    }
}

// ---------------- f32 -> f16 weight staging, zero-padded to Npad rows ----------------
__global__ void k_stage_w16(const float* __restrict__ s, _Float16* __restrict__ d,
                            int N, int K, int total /* = Npad*K */){
    int i = blockIdx.x*blockDim.x + threadIdx.x;
    if (i >= total) return;
    int n = i / K;
    d[i] = (n < N) ? (_Float16)s[i] : (_Float16)0.f;
}

// ---------------- WMMA GEMM: C(RxN) = A(RxK=96) * W(NxK=96)^T ----------------
// One wave per (16-row strip, NTILES-column group). K is compile-time so all
// per-tile B addresses fold into immediate offsets (no VGPR address pairs, no spills).
template<int NTILES>
__global__ void k_gemm_wmma(const _Float16* __restrict__ A, const _Float16* __restrict__ Wt,
                            float* __restrict__ Co, int R, int N, int colGroups)
{
    constexpr int K = DMODEL;                     // 96 for every projection here
    int gid  = (int)((blockIdx.x * blockDim.x + threadIdx.x) >> 5);
    int lane = threadIdx.x & 31;
    int rowTiles = R >> 4;
    if (gid >= rowTiles * colGroups) return;      // uniform per-wave exit
    int rt = gid / colGroups, cg = gid % colGroups;
    int row0    = rt << 4;
    int colBase = cg * (NTILES * 16);
    int m    = lane & 15;
    int half = lane >> 4;

    const _Float16* Arow  = A  + (size_t)(row0 + m)    * K + (half << 3);
    const _Float16* Bbase = Wt + (size_t)(colBase + m) * K + (half << 3);

    v8f acc[NTILES];
    #pragma unroll
    for (int nt = 0; nt < NTILES; ++nt) acc[nt] = (v8f){};

    #pragma unroll
    for (int k0 = 0; k0 < K; k0 += 32) {
        // A fragment: lanes0-15 K=k0+{0..7,16..23}; lanes16-31 K=k0+{8..15,24..31}
        v16h a;
        v8h* ah = (v8h*)&a;
        ah[0] = *(const v8h*)(Arow + k0);
        ah[1] = *(const v8h*)(Arow + k0 + 16);
        #pragma unroll
        for (int nt = 0; nt < NTILES; ++nt) {
            // B fragment: col = colBase+nt*16+m; lanes0-15 K=k0+{0..7,16..23}; lanes16-31 K=k0+{8..15,24..31}
            const _Float16* Brow = Bbase + nt * 16 * K + k0;   // compile-time offsets
            v16h bfrag;
            v8h* bh = (v8h*)&bfrag;
            bh[0] = *(const v8h*)(Brow);
            bh[1] = *(const v8h*)(Brow + 16);
            acc[nt] = __builtin_amdgcn_wmma_f32_16x16x32_f16(false, a, false, bfrag,
                                                             (short)0, acc[nt], false, false);
        }
    }
    #pragma unroll
    for (int nt = 0; nt < NTILES; ++nt) {
        int n = colBase + nt*16 + m;
        if (n < N) {
            #pragma unroll
            for (int r = 0; r < 8; ++r) {
                int mm = r + (half << 3);         // D layout: VGPR r -> M=r (+8 lanes 16-31)
                Co[(size_t)(row0 + mm) * N + n] = acc[nt][r];
            }
        }
    }
}

// ---------------- depthwise causal conv (K=4) + SiLU; rev flips the sequence ----------------
__global__ void k_conv_silu(const float* __restrict__ xz, const float* __restrict__ cw,
                            const float* __restrict__ cb, float* __restrict__ u,
                            _Float16* __restrict__ u16, int L, int rev, int total)
{
    int i = blockIdx.x*blockDim.x + threadIdx.x;
    if (i >= total) return;
    int d = i % DMODEL; int l = (i / DMODEL) % L; int b = i / (DMODEL*L);
    float acc = cb[d];
    #pragma unroll
    for (int j = 0; j < DCONV; ++j) {
        int lp = l - (DCONV-1) + j;
        if (lp >= 0) {
            int ls = rev ? (L-1-lp) : lp;
            acc += cw[d*DCONV + j] * xz[((size_t)b*L + ls)*NDBL + d];   // xp = xz[..., :96]
        }
    }
    float s = acc * sigmoidf_(acc);
    size_t oi = ((size_t)b*L + l)*DMODEL + d;
    u[oi] = s; u16[oi] = (_Float16)s;
}

// ---------------- selective scan: one thread per (b, channel), 8 states in registers -------
__global__ void k_scan(const float* __restrict__ u, const float* __restrict__ xdbl,
                       const float* __restrict__ Wdt, const float* __restrict__ bdt,
                       const float* __restrict__ Alog, const float* __restrict__ Dp,
                       float* __restrict__ y, int L, int rev)
{
    int b = blockIdx.x; int d = threadIdx.x;
    if (d >= DMODEL) return;
    float Ar[DSTATE], h[DSTATE], Wr[DTRANK];
    #pragma unroll
    for (int n = 0; n < DSTATE; ++n){ Ar[n] = -__expf(Alog[d*DSTATE + n]); h[n] = 0.f; }
    #pragma unroll
    for (int r = 0; r < DTRANK; ++r) Wr[r] = Wdt[d*DTRANK + r];
    float bd = bdt[d], dp = Dp[d];
    const float* xr = xdbl + (size_t)b*L*XDIM;
    const float* ur = u    + (size_t)b*L*DMODEL;
    float*       yr = y    + (size_t)b*L*DMODEL;
    for (int l = 0; l < L; ++l) {
        const float* xl = xr + (size_t)l*XDIM;
        __builtin_prefetch(xl + 8*XDIM, 0, 1);    // global_prefetch_b8: stream-ahead
        float dt = bd;
        #pragma unroll
        for (int r = 0; r < DTRANK; ++r) dt += xl[r]*Wr[r];
        float delta = softplusf_(dt);
        float uv = ur[(size_t)l*DMODEL + d];
        float du = delta*uv, acc = 0.f;
        #pragma unroll
        for (int n = 0; n < DSTATE; ++n) {
            float dA = __expf(delta*Ar[n]);
            h[n] = dA*h[n] + du*xl[DTRANK + n];               // B
            acc += h[n]*xl[DTRANK + DSTATE + n];              // C
        }
        int lo = rev ? (L-1-l) : l;
        yr[(size_t)lo*DMODEL + d] = acc + uv*dp;
    }
}

// ---------------- (yf+yb)/2 -> LN -> * SiLU(z) gate -> f16 ----------------
__global__ void k_combine_ln_gate(const float* __restrict__ yf, const float* __restrict__ yb,
                                  const float* __restrict__ xz, const float* __restrict__ gam,
                                  const float* __restrict__ bet, _Float16* __restrict__ g16)
{
    __shared__ float s1[128], s2[128];
    int row = blockIdx.x; int t = threadIdx.x;
    float v = 0.f;
    if (t < DMODEL) { size_t i = (size_t)row*DMODEL + t; v = 0.5f*(yf[i] + yb[i]); }
    s1[t] = v; s2[t] = v*v;
    __syncthreads();
    for (int off = 64; off > 0; off >>= 1) {
        if (t < off) { s1[t] += s1[t+off]; s2[t] += s2[t+off]; }
        __syncthreads();
    }
    float mu  = s1[0]*(1.f/DMODEL);
    float var = s2[0]*(1.f/DMODEL) - mu*mu;
    float inv = rsqrtf(var + 1e-5f);
    if (t < DMODEL) {
        float ln = (v - mu)*inv*gam[t] + bet[t];
        float z  = xz[(size_t)row*NDBL + DMODEL + t];
        g16[(size_t)row*DMODEL + t] = (_Float16)(ln * z * sigmoidf_(z));
    }
}

// ---------------- init and transposed 1/3 accumulate ----------------
__global__ void k_copy(const float* __restrict__ s, float* __restrict__ d, int n){
    int i = blockIdx.x*blockDim.x + threadIdx.x; if (i < n) d[i] = s[i];
}
__global__ void k_accum(float* __restrict__ out, const float* __restrict__ mo,
                        int mode, int Cc, int Hc, int Wc, int L, int total){
    int i = blockIdx.x*blockDim.x + threadIdx.x; if (i >= total) return;
    int w = i % Wc; int h = (i/Wc)%Hc; int c = (i/(Wc*Hc))%Cc; int b = i/(Wc*Hc*Cc);
    int row, f;
    if (mode == 1)      { row = h*Wc + w; f = c; }
    else if (mode == 2) { row = c*Wc + w; f = h; }
    else                { row = c*Hc + h; f = w; }
    out[i] += mo[((size_t)b*L + row)*DMODEL + f] * (1.f/3.f);
}

extern "C" void kernel_launch(void* const* d_in, const int* in_sizes, int n_in,
                              void* d_out, int out_size, void* d_ws, size_t ws_size,
                              hipStream_t stream)
{
    (void)in_sizes; (void)n_in; (void)out_size; (void)ws_size;
    const int Bc = 2, Cc = 96, Hc = 96, Wc = 96;
    const int L  = Hc*Wc;          // 9216 for every direction here
    const int R  = Bc*L;           // 18432 rows
    const int total = Bc*Cc*Hc*Wc;
    const int XPAD = 32;           // XDIM=22 padded to 2 column tiles
    const float* x = (const float*)d_in[0];
    float* out = (float*)d_out;

    // ---- workspace carve-out (reused across the three directions) ----
    char* ws = (char*)d_ws; size_t off = 0;
    auto carve = [&](size_t bytes)->char*{ char* p = ws + off; off += (bytes + 255) & ~(size_t)255; return p; };
    _Float16* INP16  = (_Float16*)carve((size_t)R*DMODEL*2);
    float*    XZ     = (float*)   carve((size_t)R*NDBL*4);
    float*    U0     = (float*)   carve((size_t)R*DMODEL*4);
    float*    U1     = (float*)   carve((size_t)R*DMODEL*4);
    _Float16* U016   = (_Float16*)carve((size_t)R*DMODEL*2);
    _Float16* U116   = (_Float16*)carve((size_t)R*DMODEL*2);
    float*    XD0    = (float*)   carve((size_t)R*XDIM*4);
    float*    XD1    = (float*)   carve((size_t)R*XDIM*4);
    float*    Y0     = (float*)   carve((size_t)R*DMODEL*4);
    float*    Y1     = (float*)   carve((size_t)R*DMODEL*4);
    _Float16* G16    = (_Float16*)carve((size_t)R*DMODEL*2);
    float*    MO     = (float*)   carve((size_t)R*DMODEL*4);
    _Float16* WIN16  = (_Float16*)carve((size_t)NDBL*DMODEL*2);
    _Float16* WX16F  = (_Float16*)carve((size_t)XPAD*DMODEL*2);
    _Float16* WX16B  = (_Float16*)carve((size_t)XPAD*DMODEL*2);
    _Float16* WOUT16 = (_Float16*)carve((size_t)DMODEL*DMODEL*2);

    const int rowTiles = R >> 4;                              // 1152 row strips
    auto gemm_blocks = [&](int colGroups){ return (rowTiles*colGroups*32 + 255) / 256; };

    k_copy<<<(total+255)/256, 256, 0, stream>>>(x, out, total);  // residual base

    for (int m = 0; m < 3; ++m) {
        const int mode = m + 1;
        const float* lnG = (const float*)d_in[1 + 2*m];
        const float* lnB = (const float*)d_in[2 + 2*m];
        void* const* P = d_in + 7 + 18*m;
        const float *Win=(const float*)P[0],  *cw=(const float*)P[1],   *cb=(const float*)P[2];
        const float *Wx=(const float*)P[3],   *Wdt=(const float*)P[4],  *bdt=(const float*)P[5];
        const float *Alog=(const float*)P[6], *Dp=(const float*)P[7];
        const float *cwb=(const float*)P[8],  *cbb=(const float*)P[9],  *Wxb=(const float*)P[10];
        const float *Wdtb=(const float*)P[11],*bdtb=(const float*)P[12],*Alogb=(const float*)P[13];
        const float *Dpb=(const float*)P[14], *gln=(const float*)P[15], *bln=(const float*)P[16];
        const float *Wout=(const float*)P[17];

        k_ln_transpose<<<R, 128, 0, stream>>>(x, lnG, lnB, INP16, mode, Cc, Hc, Wc, L);
        k_stage_w16<<<(NDBL*DMODEL+255)/256,   256, 0, stream>>>(Win,  WIN16,  NDBL,   DMODEL, NDBL*DMODEL);
        k_stage_w16<<<(XPAD*DMODEL+255)/256,   256, 0, stream>>>(Wx,   WX16F,  XDIM,   DMODEL, XPAD*DMODEL);
        k_stage_w16<<<(XPAD*DMODEL+255)/256,   256, 0, stream>>>(Wxb,  WX16B,  XDIM,   DMODEL, XPAD*DMODEL);
        k_stage_w16<<<(DMODEL*DMODEL+255)/256, 256, 0, stream>>>(Wout, WOUT16, DMODEL, DMODEL, DMODEL*DMODEL);

        // in-proj (WMMA): N=192 -> 12 tiles = 3 groups x NTILES=4
        k_gemm_wmma<4><<<gemm_blocks(3), 256, 0, stream>>>(INP16, WIN16, XZ, R, NDBL, 3);

        const int tconv = Bc*L*DMODEL;
        k_conv_silu<<<(tconv+255)/256, 256, 0, stream>>>(XZ, cw,  cb,  U0, U016, L, 0, tconv);
        k_conv_silu<<<(tconv+255)/256, 256, 0, stream>>>(XZ, cwb, cbb, U1, U116, L, 1, tconv);

        // x-proj (WMMA): N=22 -> 2 tiles = 1 group x NTILES=2 (padded B, guarded store)
        k_gemm_wmma<2><<<gemm_blocks(1), 256, 0, stream>>>(U016, WX16F, XD0, R, XDIM, 1);
        k_gemm_wmma<2><<<gemm_blocks(1), 256, 0, stream>>>(U116, WX16B, XD1, R, XDIM, 1);

        k_scan<<<Bc, DMODEL, 0, stream>>>(U0, XD0, Wdt,  bdt,  Alog,  Dp,  Y0, L, 0);
        k_scan<<<Bc, DMODEL, 0, stream>>>(U1, XD1, Wdtb, bdtb, Alogb, Dpb, Y1, L, 1);

        k_combine_ln_gate<<<R, 128, 0, stream>>>(Y0, Y1, XZ, gln, bln, G16);

        // out-proj (WMMA): N=96 -> 6 tiles = 2 groups x NTILES=3
        k_gemm_wmma<3><<<gemm_blocks(2), 256, 0, stream>>>(G16, WOUT16, MO, R, DMODEL, 2);

        k_accum<<<(total+255)/256, 256, 0, stream>>>(out, MO, mode, Cc, Hc, Wc, L, total);
    }
}
